// Model_27513560498946
// MI455X (gfx1250) — matrix operand order repaired and checked by reading, not compiled, not run on hardware
//
#include <hip/hip_runtime.h>
#include <hip/hip_bf16.h>

typedef __attribute__((ext_vector_type(16))) _Float16 v16h;
typedef __attribute__((ext_vector_type(8)))  float    v8f;
typedef __attribute__((ext_vector_type(4)))  float    f4;
typedef __attribute__((ext_vector_type(8)))  _Float16 h8;
typedef __attribute__((ext_vector_type(4)))  _Float16 h4;

#define TM 64
#define TN 64
#define TK 32
#define LDA_P 40   // A tile row stride (halfs); 80B rows, 16B aligned
#define LDB_P 72   // B tile transposed [TK][TN+8]; 144B rows, 16B aligned

// ---------------------------------------------------------------------------
// FAST GEMM: requires M%64==0, N%64==0, K%32==0, lda%4==0.
// C[M,N] = A[M,(lda)K] @ W[N,K]^T + bscale*bias + rscale*res
// bias/res are always valid pointers (host passes C with scale 0 when unused).
// ---------------------------------------------------------------------------
__global__ __launch_bounds__(128) void gemm_wmma_fast(
    const float* __restrict__ A, int lda,
    const float* __restrict__ W,
    const float* __restrict__ bias, float bscale,
    const float* __restrict__ res,  float rscale,
    float* __restrict__ C,
    int M, int N, int K)
{
  __shared__ __align__(16) _Float16 As[TM * LDA_P];
  __shared__ __align__(16) _Float16 Bs[TK * LDB_P];

  const int tid  = threadIdx.x;
  const int lane = tid & 31;
  const int wave = tid >> 5;
  const int wm   = (wave >> 1) * 32;
  const int wn   = (wave & 1) * 32;
  const long tileM = (long)blockIdx.y * TM;
  const long tileN = (long)blockIdx.x * TN;

  // staging coordinates (hoisted)
  const int ar  = tid >> 1;             // A row
  const int ac0 = (tid & 1) * 16;       // A col base (16 floats)
  const int bn0 = (tid & 15) * 4;       // W n-block
  const int bk0 = (tid >> 4) * 4;       // W k-block
  const float* Abase = A + (tileM + ar) * (long)lda + ac0;
  const float* Wbase = W + (tileN + bn0) * (long)K + bk0;

  v8f acc[2][2] = {};

  for (int k0 = 0; k0 < K; k0 += TK) {
    // ---- stage A (64x32): 4x float4 loads, 2x b128 LDS stores ----
    const f4* Ap = (const f4*)(Abase + k0);
    f4 a0 = Ap[0], a1 = Ap[1], a2 = Ap[2], a3 = Ap[3];
    h8 ha0, ha1;
    #pragma unroll
    for (int e = 0; e < 4; ++e) {
      ha0[e]     = (_Float16)a0[e];
      ha0[e + 4] = (_Float16)a1[e];
      ha1[e]     = (_Float16)a2[e];
      ha1[e + 4] = (_Float16)a3[e];
    }
    *(h8*)&As[ar * LDA_P + ac0]     = ha0;
    *(h8*)&As[ar * LDA_P + ac0 + 8] = ha1;

    // ---- stage W (64x32) transposed into [k][n]: 4x4 register transpose ----
    const float* Wp = Wbase + k0;
    f4 w0 = *(const f4*)(Wp);
    f4 w1 = *(const f4*)(Wp + K);
    f4 w2 = *(const f4*)(Wp + 2 * (long)K);
    f4 w3 = *(const f4*)(Wp + 3 * (long)K);
    #pragma unroll
    for (int j = 0; j < 4; ++j) {
      h4 hv;
      hv[0] = (_Float16)w0[j];
      hv[1] = (_Float16)w1[j];
      hv[2] = (_Float16)w2[j];
      hv[3] = (_Float16)w3[j];
      *(h4*)&Bs[(bk0 + j) * LDB_P + bn0] = hv;
    }
    __syncthreads();

    // ---- fragments (CDNA5 wave32 layouts) ----
    const int g  = lane >> 4;
    const int mr = lane & 15;
    v16h af[2], bf[2];
    #pragma unroll
    for (int i = 0; i < 2; ++i)
      #pragma unroll
      for (int e = 0; e < 16; ++e) {
        int kk = g * 8 + (e & 7) + ((e >> 3) << 4);
        af[i][e] = As[(wm + i * 16 + mr) * LDA_P + kk];
      }
    #pragma unroll
    for (int j = 0; j < 2; ++j)
      #pragma unroll
      for (int e = 0; e < 16; ++e) {
        int kk = g * 8 + (e & 7) + ((e >> 3) << 4);
        bf[j][e] = Bs[kk * LDB_P + (wn + j * 16 + mr)];
      }

    #pragma unroll
    for (int i = 0; i < 2; ++i)
      #pragma unroll
      for (int j = 0; j < 2; ++j)
        acc[i][j] = __builtin_amdgcn_wmma_f32_16x16x32_f16(
            false, af[i], false, bf[j], (short)0, acc[i][j], false, false);

    __syncthreads();
  }

  // ---- epilogue: single branch-free store loop ----
  const int cn = lane & 15;
  const int rb = (lane >> 4) * 8;
  const long gc0 = tileN + wn + cn;
  const long gr0 = tileM + wm + rb;
  float bv[2];
  bv[0] = bscale * bias[gc0];
  bv[1] = bscale * bias[gc0 + 16];

  #pragma unroll
  for (int j = 0; j < 2; ++j) {
    long gc = gc0 + j * 16;
    #pragma unroll
    for (int i = 0; i < 2; ++i)
      #pragma unroll
      for (int v = 0; v < 8; ++v) {
        long o = (gr0 + i * 16 + v) * (long)N + gc;
        C[o] = fmaf(rscale, res[o], acc[i][j][v] + bv[j]);
      }
  }
}

// ---------------------------------------------------------------------------
// GENERIC GEMM: any M/N/K; bias via bscale (no residual ever on this path);
// staging loops rolled, single guarded store loop with clamped addresses.
// ---------------------------------------------------------------------------
__global__ __launch_bounds__(128) void gemm_wmma_gen(
    const float* __restrict__ A, int lda,
    const float* __restrict__ W,
    const float* __restrict__ bias, float bscale,
    float* __restrict__ C,
    int M, int N, int K)
{
  __shared__ __align__(16) _Float16 As[TM * LDA_P];
  __shared__ __align__(16) _Float16 Bs[TK * LDB_P];

  const int tid  = threadIdx.x;
  const int lane = tid & 31;
  const int wave = tid >> 5;
  const int wm   = (wave >> 1) * 32;
  const int wn   = (wave & 1) * 32;
  const long tileM = (long)blockIdx.y * TM;
  const long tileN = (long)blockIdx.x * TN;

  v8f acc[2][2] = {};

  for (int k0 = 0; k0 < K; k0 += TK) {
    // clamped + selected staging, rolled loops (small static footprint)
    #pragma clang loop unroll(disable)
    for (int i = 0; i < 16; ++i) {
      int idx = tid + i * 128;
      int r = idx >> 5, c = idx & 31;
      long gr = tileM + r;
      long gk = (long)k0 + c;
      long cr = (gr < M) ? gr : (long)M - 1;
      long ck = (gk < K) ? gk : (long)K - 1;
      float av = A[cr * (long)lda + ck];
      As[r * LDA_P + c] = (_Float16)((gr < M && gk < K) ? av : 0.0f);
      long wr = tileN + r;
      long wcr = (wr < N) ? wr : (long)N - 1;
      float wv = W[wcr * (long)K + ck];
      Bs[c * LDB_P + r] = (_Float16)((wr < N && gk < K) ? wv : 0.0f);
    }
    __syncthreads();

    const int g  = lane >> 4;
    const int mr = lane & 15;
    v16h af[2], bf[2];
    #pragma unroll
    for (int i = 0; i < 2; ++i)
      #pragma unroll
      for (int e = 0; e < 16; ++e) {
        int kk = g * 8 + (e & 7) + ((e >> 3) << 4);
        af[i][e] = As[(wm + i * 16 + mr) * LDA_P + kk];
      }
    #pragma unroll
    for (int j = 0; j < 2; ++j)
      #pragma unroll
      for (int e = 0; e < 16; ++e) {
        int kk = g * 8 + (e & 7) + ((e >> 3) << 4);
        bf[j][e] = Bs[kk * LDB_P + (wn + j * 16 + mr)];
      }

    #pragma unroll
    for (int i = 0; i < 2; ++i)
      #pragma unroll
      for (int j = 0; j < 2; ++j)
        acc[i][j] = __builtin_amdgcn_wmma_f32_16x16x32_f16(
            false, af[i], false, bf[j], (short)0, acc[i][j], false, false);

    __syncthreads();
  }

  const int cn = lane & 15;
  const int rb = (lane >> 4) * 8;
  const long gc0 = tileN + wn + cn;
  const long gr0 = tileM + wm + rb;
  float bv[2];
  bv[0] = bscale * bias[(gc0      < N) ? gc0      : (long)N - 1];
  bv[1] = bscale * bias[(gc0 + 16 < N) ? gc0 + 16 : (long)N - 1];

  #pragma unroll
  for (int j = 0; j < 2; ++j) {
    long gc = gc0 + j * 16;
    #pragma unroll
    for (int i = 0; i < 2; ++i)
      #pragma unroll
      for (int v = 0; v < 8; ++v) {
        long gr = gr0 + i * 16 + v;
        float val = acc[i][j][v] + bv[j];
        if (gr < M && gc < N) C[gr * (long)N + gc] = val;
      }
  }
}

// in-place branch-free softplus (applied after dt_proj GEMM)
__global__ void softplus_kernel(float* __restrict__ d, long total)
{
  long idx = (long)blockIdx.x * blockDim.x + threadIdx.x;
  if (idx >= total) return;
  float x = d[idx];
  float sp = __logf(1.0f + __expf(fminf(x, 20.0f)));
  d[idx] = (x > 20.0f) ? x : sp;
}

// ---------------------------------------------------------------------------
// RevIN: per (b,c) mean/std over S=336, write normalized transposed (bc, s)
// ---------------------------------------------------------------------------
__global__ __launch_bounds__(128) void norm_kernel(
    const float* __restrict__ x, float* __restrict__ xn,
    float* __restrict__ mean, float* __restrict__ stdv)
{
  int bc = blockIdx.x;              // 0..63
  int b = bc >> 4, c = bc & 15;
  __shared__ float red[128];
  int tid = threadIdx.x;
  float s = 0.0f;
  for (int t = tid; t < 336; t += 128) s += x[((long)b * 336 + t) * 16 + c];
  red[tid] = s; __syncthreads();
  for (int off = 64; off > 0; off >>= 1) { if (tid < off) red[tid] += red[tid + off]; __syncthreads(); }
  float mu = red[0] / 336.0f;
  __syncthreads();
  float v = 0.0f;
  for (int t = tid; t < 336; t += 128) { float d = x[((long)b * 336 + t) * 16 + c] - mu; v += d * d; }
  red[tid] = v; __syncthreads();
  for (int off = 64; off > 0; off >>= 1) { if (tid < off) red[tid] += red[tid + off]; __syncthreads(); }
  float sd = sqrtf(red[0] / 336.0f + 1e-5f);
  if (tid == 0) { mean[bc] = mu; stdv[bc] = sd; }
  for (int t = tid; t < 336; t += 128)
    xn[(long)bc * 336 + t] = (x[((long)b * 336 + t) * 16 + c] - mu) / sd;
}

// ---------------------------------------------------------------------------
// Patch embedding (K=16) + sinusoidal positional embedding on the fly
// ---------------------------------------------------------------------------
__global__ void patch_embed_kernel(const float* __restrict__ h0,
                                   const float* __restrict__ pew,
                                   float* __restrict__ u, int total)
{
  int idx = blockIdx.x * blockDim.x + threadIdx.x;
  if (idx >= total) return;
  int dm = idx & 255;
  int t  = (idx >> 8) % 54;
  int bc = idx / (54 * 256);
  const float* hr = h0 + (long)bc * 432;
  float s = 0.0f;
  #pragma unroll
  for (int p = 0; p < 16; ++p) {
    int i = t * 8 + p; if (i > 431) i = 431;   // edge pad
    s += pew[dm * 16 + p] * hr[i];
  }
  int pair = dm >> 1;
  float freq = __expf((float)(2 * pair) * (-9.210340371976184f / 256.0f));
  float ang  = (float)t * freq;
  s += (dm & 1) ? __cosf(ang) : __sinf(ang);
  u[idx] = s;
}

// ---------------------------------------------------------------------------
// Depthwise causal conv1d (K=4) + bias + SiLU.  xi = xz[:, :di]
// ---------------------------------------------------------------------------
__global__ void conv_silu_kernel(const float* __restrict__ xz,
                                 const float* __restrict__ cw,
                                 const float* __restrict__ cb,
                                 float* __restrict__ xc, int di, int total)
{
  int idx = blockIdx.x * blockDim.x + threadIdx.x;
  if (idx >= total) return;
  int ch = idx % di;
  int l  = (idx / di) % 54;
  int bc = idx / (di * 54);
  const float* xzrow = xz + (long)bc * 54 * (2 * di);
  float s = cb[ch];
  #pragma unroll
  for (int k = 0; k < 4; ++k) {
    int li = l - 3 + k;
    if (li >= 0) s += cw[ch * 4 + k] * xzrow[(long)li * (2 * di) + ch];
  }
  xc[idx] = s / (1.0f + __expf(-s));   // SiLU
}

// ---------------------------------------------------------------------------
// Selective scan: one thread per (bc, ch), 16-state recurrence, fused z-gate
// ---------------------------------------------------------------------------
__global__ void scan_kernel(const float* __restrict__ dtb,
                            const float* __restrict__ xc,
                            const float* __restrict__ proj,
                            const float* __restrict__ xz,
                            const float* __restrict__ A_log,
                            const float* __restrict__ Dp,
                            float* __restrict__ y,
                            int di, int projN, int dtr, int total)
{
  int idx = blockIdx.x * blockDim.x + threadIdx.x;
  if (idx >= total) return;
  int ch = idx % di;
  int bc = idx / di;
  float Arow[16], h[16];
  #pragma unroll
  for (int n = 0; n < 16; ++n) { Arow[n] = -__expf(A_log[ch * 16 + n]); h[n] = 0.0f; }
  float Dv = Dp[ch];
  for (int l = 0; l < 54; ++l) {
    long row = (long)bc * 54 + l;
    float dt  = dtb[row * di + ch];
    float xv  = xc [row * di + ch];
    float dtx = dt * xv;
    const float* pr = proj + row * projN + dtr;   // [B(16) | C(16)]
    float acc = 0.0f;
    #pragma unroll
    for (int n = 0; n < 16; ++n) {
      h[n] = __expf(dt * Arow[n]) * h[n] + dtx * pr[n];
      acc += h[n] * pr[16 + n];
    }
    float yv = acc + Dv * xv;
    float z  = xz[row * (long)(2 * di) + di + ch];
    yv *= z / (1.0f + __expf(-z));                 // * silu(z)
    y[row * di + ch] = yv;
  }
}

__global__ void concat_kernel(const float* __restrict__ A, int Na,
                              const float* __restrict__ B, int Nb,
                              float* __restrict__ out, long total)
{
  long idx = (long)blockIdx.x * blockDim.x + threadIdx.x;
  if (idx >= total) return;
  int Nc = Na + Nb;
  long r = idx / Nc; int c = (int)(idx % Nc);
  out[idx] = (c < Na) ? A[r * Na + c] : B[r * Nb + (c - Na)];
}

// flat[bc, j*54+t] = h6[(bc*54+t), j]
__global__ void flat_kernel(const float* __restrict__ h6, float* __restrict__ flat)
{
  int idx = blockIdx.x * blockDim.x + threadIdx.x;
  if (idx >= 64 * 1728) return;
  int bc  = idx / 1728;
  int rem = idx % 1728;
  int j = rem / 54, t = rem % 54;
  flat[idx] = h6[((long)bc * 54 + t) * 32 + j];
}

// out[b,t,c] = ht[bc,t] * std + mean
__global__ void final_kernel(const float* __restrict__ ht,
                             const float* __restrict__ mean,
                             const float* __restrict__ stdv,
                             float* __restrict__ out)
{
  int idx = blockIdx.x * blockDim.x + threadIdx.x;
  if (idx >= 4 * 96 * 16) return;
  int c = idx % 16, t = (idx / 16) % 96, b = idx / (96 * 16);
  int bc = b * 16 + c;
  out[idx] = ht[(long)bc * 96 + t] * stdv[bc] + mean[bc];
}

// ---------------------------------------------------------------------------
struct MambaP {
  const float *in_proj, *conv_w, *conv_b, *x_proj, *dt_proj_w, *dt_proj_b,
              *A_log, *D, *out_proj;
};

extern "C" void kernel_launch(void* const* d_in, const int* in_sizes, int n_in,
                              void* d_out, int out_size, void* d_ws, size_t ws_size,
                              hipStream_t stream)
{
  (void)n_in; (void)out_size; (void)ws_size;
  auto F = [&](int i) { return (const float*)d_in[i]; };

  const float* x      = F(0);
  const float* lin0_w = F(1);  const float* lin0_b = F(2);
  const float* pe_w   = F(3);
  const float* lin1_w = F(4);  const float* lin1_b = F(5);
  const float* lin2_w = F(6);  const float* lin2_b = F(7);
  const float* lin3_w = F(8);  const float* lin3_b = F(9);
  const float* lin4_w = F(10); const float* lin4_b = F(11);
  const float* lin5_w = F(12); const float* lin5_b = F(13);
  const float* lin6_w = F(14); const float* lin6_b = F(15);
  const float* head_w = F(16); const float* head_b = F(17);

  // nested-dict flatten order probe: insertion (in_proj_w first, 65536 for m1)
  // vs sorted keys (A_log first, 4096 for m1)
  bool ins = (in_sizes[18] == 2 * 2 * 128 * 128);
  auto getM = [&](int b) {
    MambaP p;
    if (ins) {
      p.in_proj = F(b+0); p.conv_w = F(b+1); p.conv_b = F(b+2);
      p.x_proj  = F(b+3); p.dt_proj_w = F(b+4); p.dt_proj_b = F(b+5);
      p.A_log   = F(b+6); p.D = F(b+7); p.out_proj = F(b+8);
    } else {
      p.A_log = F(b+0); p.D = F(b+1); p.conv_b = F(b+2); p.conv_w = F(b+3);
      p.dt_proj_b = F(b+4); p.dt_proj_w = F(b+5); p.in_proj = F(b+6);
      p.out_proj  = F(b+7); p.x_proj = F(b+8);
    }
    return p;
  };
  MambaP m1p = getM(18), m2p = getM(27), m3p = getM(36);

  // workspace layout (floats) — total ~112 MB, L2-resident
  float* ws = (float*)d_ws;
  size_t off = 0;
  auto alloc = [&](size_t n) { float* p = ws + off; off += n; return p; };
  float* MEAN = alloc(64);
  float* STD  = alloc(64);
  float* XN   = alloc(64 * 336);
  float* H0   = alloc(64 * 432);
  float* U    = alloc(3456L * 256);
  float* H1   = alloc(3456L * 512);
  float* X3   = alloc(3456L * 512);
  float* H2   = alloc(3456L * 256);
  float* X2   = alloc(3456L * 256);
  float* H3   = alloc(3456L * 128);
  float* H3R  = alloc(3456L * 128);
  float* H4   = alloc(3456L * 256);
  float* H5   = alloc(3456L * 512);
  float* H6   = alloc(3456L * 32);
  float* FLT  = alloc(64L * 1728);
  float* HT   = alloc(64L * 96);
  float* MXZ  = alloc(3456L * 2048);   // mamba scratch sized for d=512
  float* MXC  = alloc(3456L * 1024);
  float* MPR  = alloc(3456L * 64);
  float* MDT  = alloc(3456L * 1024);
  float* MY   = alloc(3456L * 1024);
  float* CAT5 = MXZ;                   // reuse after last mamba
  float* CAT6 = MXC;                   // 3456*1024 fits exactly

  auto gemm = [&](const float* A, int lda, const float* W, const float* bias,
                  const float* res, float* C, int M, int N, int K) {
    dim3 grid((N + TN - 1) / TN, (M + TM - 1) / TM);
    bool fast = (M % TM == 0) && (N % TN == 0) && (K % TK == 0) && (lda % 4 == 0);
    const float* bp = bias ? bias : C;  float bs = bias ? 1.0f : 0.0f;
    const float* rp = res  ? res  : C;  float rs = res  ? 1.0f : 0.0f;
    if (fast)
      gemm_wmma_fast<<<grid, 128, 0, stream>>>(A, lda, W, bp, bs, rp, rs, C, M, N, K);
    else
      gemm_wmma_gen<<<grid, 128, 0, stream>>>(A, lda, W, bp, bs, C, M, N, K);
  };

  auto run_mamba = [&](const float* hin, float* hout, const MambaP& p, int d) {
    int di = 2 * d;
    int dtr = (d + 15) / 16;       // 32 / 16 / 8
    int projN = dtr + 32;
    const int M = 3456;
    gemm(hin, d, p.in_proj, nullptr, nullptr, MXZ, M, 2 * di, d);
    int tc = 64 * 54 * di;
    conv_silu_kernel<<<(tc + 255) / 256, 256, 0, stream>>>(MXZ, p.conv_w, p.conv_b, MXC, di, tc);
    gemm(MXC, di, p.x_proj, nullptr, nullptr, MPR, M, projN, di);
    gemm(MPR, projN, p.dt_proj_w, p.dt_proj_b, nullptr, MDT, M, di, dtr);
    long tsp = (long)M * di;
    softplus_kernel<<<(unsigned)((tsp + 255) / 256), 256, 0, stream>>>(MDT, tsp);
    int ts = 64 * di;
    scan_kernel<<<(ts + 255) / 256, 256, 0, stream>>>(MDT, MXC, MPR, MXZ, p.A_log, p.D,
                                                      MY, di, projN, dtr, ts);
    gemm(MY, di, p.out_proj, nullptr, hin, hout, M, d, di);                   // + residual
  };

  // --- pipeline ---
  norm_kernel<<<64, 128, 0, stream>>>(x, XN, MEAN, STD);
  gemm(XN, 336, lin0_w, lin0_b, nullptr, H0, 64, 432, 336);

  int tp = 3456 * 256;
  patch_embed_kernel<<<(tp + 255) / 256, 256, 0, stream>>>(H0, pe_w, U, tp);

  gemm(U, 256, lin1_w, lin1_b, nullptr, H1, 3456, 512, 256);
  run_mamba(H1, X3, m3p, 512);

  gemm(H1, 512, lin2_w, lin2_b, nullptr, H2, 3456, 256, 512);
  run_mamba(H2, X2, m2p, 256);

  gemm(H2, 256, lin3_w, lin3_b, nullptr, H3, 3456, 128, 256);
  run_mamba(H3, H3R, m1p, 128);

  gemm(H3R, 128, lin4_w, lin4_b, nullptr, H4, 3456, 256, 128);

  long t5 = 3456L * 512;
  concat_kernel<<<(unsigned)((t5 + 255) / 256), 256, 0, stream>>>(H4, 256, X2, 256, CAT5, t5);
  gemm(CAT5, 512, lin5_w, lin5_b, nullptr, H5, 3456, 512, 512);

  long t6 = 3456L * 1024;
  concat_kernel<<<(unsigned)((t6 + 255) / 256), 256, 0, stream>>>(H5, 512, X3, 512, CAT6, t6);
  gemm(CAT6, 1024, lin6_w, lin6_b, nullptr, H6, 3456, 32, 1024);

  flat_kernel<<<(64 * 1728 + 255) / 256, 256, 0, stream>>>(H6, FLT);
  gemm(FLT, 1728, head_w, head_b, nullptr, HT, 64, 96, 1728);
  final_kernel<<<(6144 + 255) / 256, 256, 0, stream>>>(HT, MEAN, STD, (float*)d_out);
}
